// MultiHeadedAttention_63230508532529
// MI455X (gfx1250) — matrix operand order repaired and checked by reading, not compiled, not run on hardware
//
#include <hip/hip_runtime.h>
#include <hip/hip_bf16.h>

// ---------------------------------------------------------------------------
// MultiHeadedAttention on MI455X (gfx1250): bf16 WMMA + async LDS staging.
//   B=4, S=T=2048, D=1024, H=16, HW=64
// ---------------------------------------------------------------------------

typedef __attribute__((ext_vector_type(16))) __bf16 v16bf;
typedef __attribute__((ext_vector_type(8)))  float  v8f;
typedef int v4i __attribute__((vector_size(16)));   // matches builtin param type

#define B_  4
#define S_  2048
#define T_  2048
#define D_  1024
#define H_  16
#define HW_ 64

#ifndef __has_builtin
#define __has_builtin(x) 0
#endif
#if __has_builtin(__builtin_amdgcn_global_load_async_to_lds_b128)
#define HAVE_ASYNC_LDS 1
#else
#define HAVE_ASYNC_LDS 0
#endif

union Frag { v16bf v; uint4 u[2]; };          // 32 bytes: one A/B WMMA operand
union Acc  { v8f v; float f[8]; };            // 32 bytes: one 16x16 f32 C/D tile

__device__ __forceinline__ unsigned short f2bf(float f) {
    unsigned int u = __float_as_uint(f);
    unsigned int r = (u + 0x7FFFu + ((u >> 16) & 1u)) >> 16;   // RNE
    return (unsigned short)r;
}

__device__ __forceinline__ v8f wmma_bf16(v16bf a, v16bf b, v8f c) {
    return __builtin_amdgcn_wmma_f32_16x16x32_bf16(
        /*neg_a=*/false, a, /*neg_b=*/false, b,
        /*c_mod=*/(short)0, c, /*reuse_a=*/false, /*reuse_b=*/false);
}

// 16-byte global->LDS copy: async (ASYNCcnt-tracked) on gfx1250.
__device__ __forceinline__ void copy16_g2l(const void* g, void* l) {
#if HAVE_ASYNC_LDS
    __builtin_amdgcn_global_load_async_to_lds_b128(
        (__attribute__((address_space(1))) v4i*)g,
        (__attribute__((address_space(3))) v4i*)l, 0, 0);
#else
    *(uint4*)l = *(const uint4*)g;
#endif
}

template <int N>
__device__ __forceinline__ void wait_async() {
#if HAVE_ASYNC_LDS
#if __has_builtin(__builtin_amdgcn_s_wait_asynccnt)
    __builtin_amdgcn_s_wait_asynccnt((unsigned short)N);
#else
    asm volatile("s_wait_asynccnt %0" :: "i"(N) : "memory");
#endif
#endif
}

// ---------------------------------------------------------------------------
// Elementwise fp32 -> bf16 (8 elements / thread).
// ---------------------------------------------------------------------------
__global__ __launch_bounds__(256) void cvt_bf16_kernel(
    const float* __restrict__ src, unsigned short* __restrict__ dst, int n8)
{
    const int i = blockIdx.x * blockDim.x + threadIdx.x;
    if (i >= n8) return;
    union { float4 v; float f[4]; } a, b;
    a.v = ((const float4*)src)[2 * (size_t)i];
    b.v = ((const float4*)src)[2 * (size_t)i + 1];
    uint4 o;
    o.x = (unsigned)f2bf(a.f[0]) | ((unsigned)f2bf(a.f[1]) << 16);
    o.y = (unsigned)f2bf(a.f[2]) | ((unsigned)f2bf(a.f[3]) << 16);
    o.z = (unsigned)f2bf(b.f[0]) | ((unsigned)f2bf(b.f[1]) << 16);
    o.w = (unsigned)f2bf(b.f[2]) | ((unsigned)f2bf(b.f[3]) << 16);
    ((uint4*)dst)[i] = o;
}

// ---------------------------------------------------------------------------
// Weight transpose + convert: W fp32 [K][N] -> Wt bf16 [N][K].
// ---------------------------------------------------------------------------
__global__ __launch_bounds__(256) void wtrans_bf16_kernel(
    const float* __restrict__ W, unsigned short* __restrict__ Wt, int K, int N)
{
    __shared__ unsigned short tile[32][33];
    const int tx = threadIdx.x & 31;
    const int ty = threadIdx.x >> 5;          // 0..7
    const int k0 = blockIdx.y * 32, n0 = blockIdx.x * 32;
#pragma unroll
    for (int r = ty; r < 32; r += 8)
        tile[r][tx] = f2bf(W[(size_t)(k0 + r) * N + n0 + tx]);
    __syncthreads();
#pragma unroll
    for (int r = ty; r < 32; r += 8)
        Wt[(size_t)(n0 + r) * K + k0 + tx] = tile[tx][r];
}

// ---------------------------------------------------------------------------
// Pure-bf16 WMMA GEMM: Out = A[MxK] * Wt[NxK]^T + bias
//   A bf16 row-major, Wt bf16 transposed (col-major weights).
//   Double-buffered LDS tiles filled with async b128 copies.
// MODE 0: out bf16 -> q/k layout  [b,h,s,64]
// MODE 1: out bf16 -> v transpose [b,h,64,t]
// MODE 2: out fp32 -> plain row-major (final projection)
// Block: 256 thr = 8 waves (2 along M x 4 along N); block tile 128x128.
// ---------------------------------------------------------------------------
template <int MODE>
__global__ __launch_bounds__(256) void gemm_bf16_kernel(
    const unsigned short* __restrict__ A, const unsigned short* __restrict__ Wt,
    const float* __restrict__ bias, void* __restrict__ Out,
    int M, int K, int N)
{
    __shared__ unsigned short lds_a[2][128 * 32];   // [row][k] row-major
    __shared__ unsigned short lds_b[2][128 * 32];   // [col][k] col-major

    const int t    = threadIdx.x;
    const int lane = t & 31;
    const int wave = t >> 5;
    const int wm   = wave >> 2;       // 0..1 : M half
    const int wn   = wave & 3;        // 0..3 : N quarter
    const int lhi  = lane >> 4;
    const int llo  = lane & 15;
    const int m_blk = blockIdx.y * 128;
    const int n_blk = blockIdx.x * 128;

    Acc acc[4][2];
#pragma unroll
    for (int mt = 0; mt < 4; ++mt)
#pragma unroll
        for (int nt = 0; nt < 2; ++nt)
#pragma unroll
            for (int j = 0; j < 8; ++j) acc[mt][nt].f[j] = 0.0f;

    const int nK = K >> 5;

    // Tile stager: 512 16-byte chunks per operand tile; 2 A + 2 B per thread.
    auto stage = [&](int kt, int buf) {
#pragma unroll
        for (int c = t; c < 512; c += 256) {
            const int rr = c >> 2;
            const int pp = (c & 3) * 8;
            copy16_g2l(A  + (size_t)(m_blk + rr) * K + kt * 32 + pp,
                       &lds_a[buf][rr * 32 + pp]);
            copy16_g2l(Wt + (size_t)(n_blk + rr) * K + kt * 32 + pp,
                       &lds_b[buf][rr * 32 + pp]);
        }
    };

    stage(0, 0);

    for (int kt = 0; kt < nK; ++kt) {
        const int buf = kt & 1;
        if (kt + 1 < nK) {
            stage(kt + 1, buf ^ 1);
            wait_async<4>();          // current tile's copies done; next in flight
        } else {
            wait_async<0>();
        }
        __syncthreads();

        Frag af[4], bfr[2];
#pragma unroll
        for (int mt = 0; mt < 4; ++mt) {
            const int row = wm * 64 + mt * 16 + llo;
            af[mt].u[0] = *(const uint4*)&lds_a[buf][row * 32 + 8 * lhi];
            af[mt].u[1] = *(const uint4*)&lds_a[buf][row * 32 + 16 + 8 * lhi];
        }
#pragma unroll
        for (int nt = 0; nt < 2; ++nt) {
            const int col = wn * 32 + nt * 16 + llo;
            const uint4* p = (const uint4*)&lds_b[buf][col * 32 + 8 * lhi];
            bfr[nt].u[0] = p[0];
            bfr[nt].u[1] = p[2];
        }
#pragma unroll
        for (int mt = 0; mt < 4; ++mt)
#pragma unroll
            for (int nt = 0; nt < 2; ++nt)
                acc[mt][nt].v = wmma_bf16(af[mt].v, bfr[nt].v, acc[mt][nt].v);
        __syncthreads();              // all reads done before buf is re-staged
    }

    // ---- epilogue ----
#pragma unroll
    for (int mt = 0; mt < 4; ++mt)
#pragma unroll
        for (int nt = 0; nt < 2; ++nt)
#pragma unroll
            for (int j = 0; j < 8; ++j) {
                const int row = m_blk + wm * 64 + mt * 16 + j + 8 * lhi;
                const int col = n_blk + wn * 32 + nt * 16 + llo;
                float v = acc[mt][nt].f[j] + bias[col];
                if (MODE == 0) {
                    const int b = row >> 11, s = row & 2047;
                    const int h = col >> 6,  d = col & 63;
                    ((unsigned short*)Out)[(((size_t)(b * H_ + h)) * S_ + s) * HW_ + d] = f2bf(v);
                } else if (MODE == 1) {
                    const int b = row >> 11, s = row & 2047;
                    const int h = col >> 6,  d = col & 63;
                    ((unsigned short*)Out)[(((size_t)(b * H_ + h)) * HW_ + d) * T_ + s] = f2bf(v);
                } else {
                    ((float*)Out)[(size_t)row * N + col] = v;
                }
            }
}

// ---------------------------------------------------------------------------
// Flash attention per (b,h): grid (S/64, B*H), 128 threads = 4 waves.
// Each wave owns 16 query rows; loops T in 64-token chunks.
//   q: [b,h,s,64] bf16   k: [b,h,t,64] bf16   v_t: [b,h,64,t] bf16
//   attn out: [b,s,h,64] bf16 (A-operand layout for the final GEMM)
// Mask tile (64x64 bytes) staged to LDS with async copies, overlapped with
// the Q*K^T WMMAs.
// ---------------------------------------------------------------------------
__global__ __launch_bounds__(128) void flash_attn_kernel(
    const unsigned short* __restrict__ qb,
    const unsigned short* __restrict__ kb,
    const unsigned short* __restrict__ vtb,
    const unsigned char*  __restrict__ mask,
    unsigned short*       __restrict__ attn)
{
    __shared__ unsigned short plds[4][16 * 64];   // per-wave P tile (bf16)
#if HAVE_ASYNC_LDS
    __shared__ unsigned char  mlds[64 * 64];      // mask tile for this block
#endif

    const int lane = threadIdx.x & 31;
    const int wave = threadIdx.x >> 5;
    const int lhi  = lane >> 4;
    const int llo  = lane & 15;

    const int bh = blockIdx.y;
    const int b  = bh >> 4;
    const int h  = bh & 15;
    const int sblk = blockIdx.x * 64;
    const int s0   = sblk + wave * 16;

    const unsigned short* qbase = qb  + ((size_t)bh * S_ + s0) * HW_;
    const unsigned short* kbase = kb  + (size_t)bh * T_ * HW_;
    const unsigned short* vbase = vtb + (size_t)bh * HW_ * T_;
    const unsigned char*  mbase = mask + (size_t)b * S_ * T_;
    unsigned short* myp = plds[wave];

    // Q A-fragments for the two 32-wide K-chunks of the 64-dim head.
    Frag qf[2];
#pragma unroll
    for (int kc = 0; kc < 2; ++kc) {
        const unsigned short* p = qbase + (size_t)llo * HW_ + kc * 32 + 8 * lhi;
        qf[kc].u[0] = *(const uint4*)p;
        qf[kc].u[1] = *(const uint4*)(p + 16);
    }

    float run_m[8], run_l[8];
    Acc ao[4];
#pragma unroll
    for (int j = 0; j < 8; ++j) { run_m[j] = -3.0e38f; run_l[j] = 0.0f; }
#pragma unroll
    for (int nt = 0; nt < 4; ++nt)
#pragma unroll
        for (int j = 0; j < 8; ++j) ao[nt].f[j] = 0.0f;

    for (int jt = 0; jt < T_; jt += 64) {
#if HAVE_ASYNC_LDS
        // ---- stage mask tile (64x64 u8) asynchronously; 32B per thread ----
        {
            const int mr = threadIdx.x >> 1;
            const int mo = (threadIdx.x & 1) * 32;
            const unsigned char* msrc = mbase + (size_t)(sblk + mr) * T_ + jt + mo;
            unsigned char* mdst = &mlds[mr * 64 + mo];
            copy16_g2l(msrc,      mdst);
            copy16_g2l(msrc + 16, mdst + 16);
        }
#endif
        // ---- prefetch next chunk of K / V ----
        if (jt + 64 < T_) {
            __builtin_prefetch(kbase + (size_t)(jt + 64 + 2 * lane) * HW_, 0, 0);
            __builtin_prefetch(vbase + (size_t)(2 * lane) * T_ + jt + 64, 0, 0);
        }

        // ---- S = Q * K^T (16 x 64), 8 WMMAs ----
        Acc sa[4];
#pragma unroll
        for (int nt = 0; nt < 4; ++nt)
#pragma unroll
            for (int j = 0; j < 8; ++j) sa[nt].f[j] = 0.0f;

#pragma unroll
        for (int nt = 0; nt < 4; ++nt) {
            const int tok = jt + nt * 16 + llo;
#pragma unroll
            for (int kc = 0; kc < 2; ++kc) {
                Frag kf;
                const uint4* p = (const uint4*)(kbase + (size_t)tok * HW_ + kc * 32 + 8 * lhi);
                kf.u[0] = p[0];
                kf.u[1] = p[2];
                sa[nt].v = wmma_bf16(qf[kc].v, kf.v, sa[nt].v);
            }
        }

        // ---- mask ----
#if HAVE_ASYNC_LDS
        wait_async<0>();
        __syncthreads();
#pragma unroll
        for (int nt = 0; nt < 4; ++nt)
#pragma unroll
            for (int j = 0; j < 8; ++j) {
                const int rrow = wave * 16 + j + 8 * lhi;
                const int tcol = nt * 16 + llo;
                if (mlds[rrow * 64 + tcol]) sa[nt].f[j] = -100000000.0f;
            }
#else
#pragma unroll
        for (int nt = 0; nt < 4; ++nt)
#pragma unroll
            for (int j = 0; j < 8; ++j) {
                const int srow = s0 + j + 8 * lhi;
                const int tcol = jt + nt * 16 + llo;
                if (mbase[(size_t)srow * T_ + tcol]) sa[nt].f[j] = -100000000.0f;
            }
#endif

        // ---- online softmax (rows live across 16-lane halves) ----
        float nm[8], alpha[8];
#pragma unroll
        for (int j = 0; j < 8; ++j) {
            float m = fmaxf(fmaxf(sa[0].f[j], sa[1].f[j]), fmaxf(sa[2].f[j], sa[3].f[j]));
            m = fmaxf(m, __shfl_xor(m, 1, 32));
            m = fmaxf(m, __shfl_xor(m, 2, 32));
            m = fmaxf(m, __shfl_xor(m, 4, 32));
            m = fmaxf(m, __shfl_xor(m, 8, 32));
            nm[j]    = fmaxf(run_m[j], m);
            alpha[j] = __expf(run_m[j] - nm[j]);
            run_m[j] = nm[j];
        }
#pragma unroll
        for (int j = 0; j < 8; ++j) {
            float s = 0.0f;
#pragma unroll
            for (int nt = 0; nt < 4; ++nt) {
                float p = __expf(sa[nt].f[j] - nm[j]);
                sa[nt].f[j] = p;
                s += p;
            }
            s += __shfl_xor(s, 1, 32);
            s += __shfl_xor(s, 2, 32);
            s += __shfl_xor(s, 4, 32);
            s += __shfl_xor(s, 8, 32);
            run_l[j] = run_l[j] * alpha[j] + s;
        }
#pragma unroll
        for (int nt = 0; nt < 4; ++nt)
#pragma unroll
            for (int j = 0; j < 8; ++j) ao[nt].f[j] *= alpha[j];

        // ---- P: C-layout -> A-fragment layout via LDS ----
        __syncthreads();   // WAR: previous iteration's P reads are done
#pragma unroll
        for (int nt = 0; nt < 4; ++nt)
#pragma unroll
            for (int j = 0; j < 8; ++j)
                myp[(j + 8 * lhi) * 64 + nt * 16 + llo] = f2bf(sa[nt].f[j]);
        __syncthreads();   // RAW: all lanes' P stores visible

        Frag pf[2];
#pragma unroll
        for (int kc = 0; kc < 2; ++kc) {
            pf[kc].u[0] = *(const uint4*)&myp[llo * 64 + kc * 32 + 8 * lhi];
            pf[kc].u[1] = *(const uint4*)&myp[llo * 64 + kc * 32 + 16 + 8 * lhi];
        }

        // ---- O += P * V (16 x 64), 8 WMMAs; V is token-contiguous per dim ----
#pragma unroll
        for (int nt = 0; nt < 4; ++nt) {
            const int d = nt * 16 + llo;
#pragma unroll
            for (int kc = 0; kc < 2; ++kc) {
                Frag vf;
                const uint4* p = (const uint4*)(vbase + (size_t)d * T_ + jt + kc * 32 + 8 * lhi);
                vf.u[0] = p[0];
                vf.u[1] = p[2];
                ao[nt].v = wmma_bf16(pf[kc].v, vf.v, ao[nt].v);
            }
        }
    }

    // ---- normalize: 1/row_sum and post-softmax 1/sqrt(64) ----
#pragma unroll
    for (int j = 0; j < 8; ++j) run_l[j] = 1.0f / (run_l[j] * 8.0f);
#pragma unroll
    for (int nt = 0; nt < 4; ++nt)
#pragma unroll
        for (int j = 0; j < 8; ++j) {
            const int srow = s0 + j + 8 * lhi;
            const int d    = nt * 16 + llo;
            attn[(((size_t)(b * S_ + srow)) * H_ + h) * HW_ + d] =
                f2bf(ao[nt].f[j] * run_l[j]);
        }
}

// ---------------------------------------------------------------------------
extern "C" void kernel_launch(void* const* d_in, const int* in_sizes, int n_in,
                              void* d_out, int out_size, void* d_ws, size_t ws_size,
                              hipStream_t stream) {
    (void)in_sizes; (void)n_in; (void)out_size; (void)ws_size;
    const float* query = (const float*)d_in[0];
    const float* key   = (const float*)d_in[1];
    const float* value = (const float*)d_in[2];
    const unsigned char* mask = (const unsigned char*)d_in[3];
    const float* Wq = (const float*)d_in[4];
    const float* bq = (const float*)d_in[5];
    const float* Wk = (const float*)d_in[6];
    const float* bk = (const float*)d_in[7];
    const float* Wv = (const float*)d_in[8];
    const float* bv = (const float*)d_in[9];
    const float* Wo = (const float*)d_in[10];
    const float* bo = (const float*)d_in[11];
    float* out = (float*)d_out;

    const size_t e  = (size_t)B_ * H_ * S_ * HW_;   // 8,388,608 elems
    const size_t we = (size_t)D_ * D_;              // 1,048,576 elems
    unsigned short* qb   = (unsigned short*)d_ws;   // bf16 workspace buffers
    unsigned short* kb   = qb   + e;
    unsigned short* vtb  = kb   + e;
    unsigned short* attn = vtb  + e;
    unsigned short* qin  = attn + e;                // bf16 copies of inputs
    unsigned short* kin  = qin  + e;
    unsigned short* vin  = kin  + e;
    unsigned short* wq_t = vin  + e;                // bf16 transposed weights
    unsigned short* wk_t = wq_t + we;
    unsigned short* wv_t = wk_t + we;
    unsigned short* wo_t = wv_t + we;

    const int M = B_ * S_;                          // 8192

    // 1) convert activations fp32 -> bf16 (row-major, once)
    {
        const int n8 = (int)(e / 8);                 // 1,048,576
        dim3 cg((n8 + 255) / 256);
        cvt_bf16_kernel<<<cg, 256, 0, stream>>>(query, qin, n8);
        cvt_bf16_kernel<<<cg, 256, 0, stream>>>(key,   kin, n8);
        cvt_bf16_kernel<<<cg, 256, 0, stream>>>(value, vin, n8);
    }
    // 2) convert + transpose weights fp32 [K][N] -> bf16 [N][K]
    {
        dim3 wg(D_ / 32, D_ / 32);
        wtrans_bf16_kernel<<<wg, 256, 0, stream>>>(Wq, wq_t, D_, D_);
        wtrans_bf16_kernel<<<wg, 256, 0, stream>>>(Wk, wk_t, D_, D_);
        wtrans_bf16_kernel<<<wg, 256, 0, stream>>>(Wv, wv_t, D_, D_);
        wtrans_bf16_kernel<<<wg, 256, 0, stream>>>(Wo, wo_t, D_, D_);
    }

    dim3 ggrid(D_ / 128, M / 128);                  // (8, 64)
    gemm_bf16_kernel<0><<<ggrid, 256, 0, stream>>>(qin,  wq_t, bq, qb,  M, D_, D_);
    gemm_bf16_kernel<0><<<ggrid, 256, 0, stream>>>(kin,  wk_t, bk, kb,  M, D_, D_);
    gemm_bf16_kernel<1><<<ggrid, 256, 0, stream>>>(vin,  wv_t, bv, vtb, M, D_, D_);

    flash_attn_kernel<<<dim3(S_ / 64, B_ * H_), 128, 0, stream>>>(qb, kb, vtb, mask, attn);

    gemm_bf16_kernel<2><<<ggrid, 256, 0, stream>>>(attn, wo_t, bo, out, M, D_, D_);
}
